// StableCrossAttention_44925357916703
// MI455X (gfx1250) — compile-verified, matrix operand order checked by reading, not run on hardware
//
#include <hip/hip_runtime.h>
#include <hip/hip_bf16.h>
#include <stdint.h>

// ---------------------------------------------------------------------------
// StableCrossAttention on MI455X (gfx1250):
//   - all GEMMs via v_wmma_f32_16x16x32_bf16
//   - one-time fp32->bf16 conversion pass so GEMM staging is pure copy
//   - double-buffered LDS tiles filled with GLOBAL_LOAD_ASYNC_TO_LDS_B128
//     (ASYNCcnt-tracked; s_wait_asynccnt 2 => previous tile complete)
//   - softmax row reductions via DS_SWIZZLE_B32 immediate XOR patterns
// ---------------------------------------------------------------------------

typedef __attribute__((ext_vector_type(16))) __bf16 v16bf;
typedef __attribute__((ext_vector_type(8)))  float  v8f;

union AFrag { v16bf v; unsigned int u[8]; };
union CFrag { v8f   v; float        f[8]; };

#define HIDDEN  1024
#define NHEADS  16
#define HDIM    64

__device__ __forceinline__ unsigned short f2bf(float x) {
    unsigned int u = __float_as_uint(x);
    u += 0x7FFFu + ((u >> 16) & 1u);
    return (unsigned short)(u >> 16);
}
__device__ __forceinline__ unsigned int packbf2(float lo, float hi) {
    return (unsigned int)f2bf(lo) | ((unsigned int)f2bf(hi) << 16);
}
__device__ __forceinline__ float clampf(float x, float c) {
    return fminf(fmaxf(x, -c), c);
}
// 16-bit A-matrix 16x32 K-pair index for (vgpr i, lane-half)
__device__ __forceinline__ int pairmap(int i, int half) {
    return (i < 4) ? (half * 4 + i) : (8 + half * 4 + (i - 4));
}

// ds_swizzle group-of-32 XOR butterfly: lane' = (lane & 0x1F) ^ xor_mask.
// XOR masks {1,2,4,8} stay inside each 16-lane half (rows of a C-fragment
// live in one half), so these implement width-16 reductions with zero
// address-VALU overhead.
template <int XORM>
__device__ __forceinline__ float swz_xor(float v) {
    int r = __builtin_amdgcn_ds_swizzle(__float_as_int(v), 0x1F | (XORM << 10));
    return __int_as_float(r);
}
__device__ __forceinline__ float redmax16(float v) {
    v = fmaxf(v, swz_xor<8>(v));
    v = fmaxf(v, swz_xor<4>(v));
    v = fmaxf(v, swz_xor<2>(v));
    v = fmaxf(v, swz_xor<1>(v));
    return v;
}
__device__ __forceinline__ float redsum16(float v) {
    v += swz_xor<8>(v);
    v += swz_xor<4>(v);
    v += swz_xor<2>(v);
    v += swz_xor<1>(v);
    return v;
}

// generic-pointer low 32 bits == LDS byte offset
__device__ __forceinline__ unsigned lds_off(const void* p) {
    return (unsigned)(uintptr_t)p;
}
// per-lane async copy of 16B global -> 16B LDS (ASYNCcnt tracked)
__device__ __forceinline__ void async_b128(unsigned lds_addr, unsigned long long gaddr) {
    asm volatile("global_load_async_to_lds_b128 %0, %1, off"
                 :: "v"(lds_addr), "v"(gaddr) : "memory");
}
__device__ __forceinline__ void wait_async2() {
    asm volatile("s_wait_asynccnt 0x2" ::: "memory");
}
__device__ __forceinline__ void wait_async0() {
    asm volatile("s_wait_asynccnt 0x0" ::: "memory");
}

// ---------------------------------------------------------------------------
// fp32 -> bf16(clamp 50) conversion pass. n_pairs = elems/2.
// ---------------------------------------------------------------------------
__global__ __launch_bounds__(256) void conv_kernel(const float* __restrict__ x,
                                                   unsigned int* __restrict__ y,
                                                   int n_pairs) {
    int base = (blockIdx.x * 256 + threadIdx.x) * 4;
    const float2* x2 = (const float2*)x;
#pragma unroll
    for (int i = 0; i < 4; ++i) {
        int idx = base + i;
        if (idx < n_pairs) {
            float2 v = x2[idx];
            y[idx] = packbf2(clampf(v.x, 50.0f), clampf(v.y, 50.0f));
        }
    }
}

// ---------------------------------------------------------------------------
// Projection: Y[m,n] = clamp( Xbf[m,:] . Wbf[n,:] + b[n], 50 )
//   OUT_MODE: 0 = bf16 row-major [M][1024]
//             1 = bf16 per-head transposed [b][h][d][2048]   (V path)
//             2 = fp32 row-major [M][1024]                   (final output)
// Block: 256 threads = 8 waves; 64x64 tile; K staged via async b128, double buf.
// ---------------------------------------------------------------------------
template <int OUT_MODE>
__global__ __launch_bounds__(256) void proj_kernel(const unsigned short* __restrict__ Xb,
                                                   const unsigned short* __restrict__ Wb,
                                                   const float* __restrict__ bias,
                                                   void* __restrict__ Yv,
                                                   int M) {
    __shared__ unsigned int Xl[2][64 * 17];   // 64 rows x 16 k-pairs (padded)
    __shared__ unsigned int Wl[2][64 * 17];

    const int t     = threadIdx.x;
    const int w     = t >> 5;
    const int lane  = t & 31;
    const int lm    = lane & 15;
    const int half  = lane >> 4;
    const int mbase = blockIdx.x * 64;
    const int nbase = blockIdx.y * 64;
    const int mtile = w & 3;
    const int npair = w >> 2;

    // per-lane staging coords: one b128 (4 pair-uints) per lane per tile
    const int srow = t >> 2;          // 0..63
    const int spg  = (t & 3) * 4;     // pair-uint group 0/4/8/12

    const unsigned long long Xg = (unsigned long long)(uintptr_t)Xb;
    const unsigned long long Wg = (unsigned long long)(uintptr_t)Wb;

    CFrag c0, c1;
#pragma unroll
    for (int i = 0; i < 8; ++i) { c0.f[i] = 0.0f; c1.f[i] = 0.0f; }

    // prologue: stage k-step 0 into buffer 0
    {
        async_b128(lds_off(&Xl[0][srow * 17 + spg]),
                   Xg + ((size_t)(mbase + srow) * HIDDEN + 0 + spg * 2) * 2);
        async_b128(lds_off(&Wl[0][srow * 17 + spg]),
                   Wg + ((size_t)(nbase + srow) * HIDDEN + 0 + spg * 2) * 2);
    }

    for (int step = 0; step < 32; ++step) {
        if (step + 1 < 32) {
            int kk = (step + 1) * 32;
            int s  = (step + 1) & 1;
            async_b128(lds_off(&Xl[s][srow * 17 + spg]),
                       Xg + ((size_t)(mbase + srow) * HIDDEN + kk + spg * 2) * 2);
            async_b128(lds_off(&Wl[s][srow * 17 + spg]),
                       Wg + ((size_t)(nbase + srow) * HIDDEN + kk + spg * 2) * 2);
            wait_async2();   // loads retire in order: current step's tile landed
        } else {
            wait_async0();
        }
        __syncthreads();

        const unsigned int* Xc = Xl[step & 1];
        const unsigned int* Wc = Wl[step & 1];
        AFrag A, B0, B1;
#pragma unroll
        for (int i = 0; i < 8; ++i) {
            int pm = pairmap(i, half);
            A.u[i]  = Xc[(mtile * 16 + lm) * 17 + pm];
            B0.u[i] = Wc[(npair * 32 + lm) * 17 + pm];
            B1.u[i] = Wc[(npair * 32 + 16 + lm) * 17 + pm];
        }
        c0.v = __builtin_amdgcn_wmma_f32_16x16x32_bf16(false, A.v, false, B0.v,
                                                       (short)0, c0.v, false, false);
        c1.v = __builtin_amdgcn_wmma_f32_16x16x32_bf16(false, A.v, false, B1.v,
                                                       (short)0, c1.v, false, false);
        __syncthreads();
    }

    const int ncol0 = nbase + npair * 32 + lm;
    const int ncol1 = ncol0 + 16;
    const float bv0 = bias[ncol0];
    const float bv1 = bias[ncol1];
#pragma unroll
    for (int i = 0; i < 8; ++i) {
        int mrow = mbase + mtile * 16 + half * 8 + i;
        float y0 = clampf(c0.f[i] + bv0, 50.0f);
        float y1 = clampf(c1.f[i] + bv1, 50.0f);
        if (OUT_MODE == 0) {
            unsigned short* Y = (unsigned short*)Yv;
            Y[(size_t)mrow * HIDDEN + ncol0] = f2bf(y0);
            Y[(size_t)mrow * HIDDEN + ncol1] = f2bf(y1);
        } else if (OUT_MODE == 1) {
            unsigned short* Y = (unsigned short*)Yv;   // [b][h][d][2048]
            int bb  = mrow >> 11;
            int tok = mrow & 2047;
            int h0 = ncol0 >> 6, d0 = ncol0 & 63;
            int h1 = ncol1 >> 6, d1 = ncol1 & 63;
            Y[((((size_t)bb * NHEADS + h0) * HDIM) + d0) * 2048 + tok] = f2bf(y0);
            Y[((((size_t)bb * NHEADS + h1) * HDIM) + d1) * 2048 + tok] = f2bf(y1);
        } else {
            float* Y = (float*)Yv;
            Y[(size_t)mrow * HIDDEN + ncol0] = y0;
            Y[(size_t)mrow * HIDDEN + ncol1] = y1;
        }
    }
    (void)M;
}

// ---------------------------------------------------------------------------
// Flash attention: block = (b, h, 128 q rows); 8 waves x 16-row q tiles.
// Double-buffered K chunk (32 keys x 64d) and transposed-V chunk (64d x 32 keys)
// staged with async b128; online softmax; P re-layout via per-wave LDS scratch.
// ---------------------------------------------------------------------------
__global__ __launch_bounds__(256) void attn_kernel(const unsigned int* __restrict__ Qb,
                                                   const unsigned int* __restrict__ Kb,
                                                   const unsigned int* __restrict__ Vt,
                                                   const unsigned char* __restrict__ mask,
                                                   unsigned short* __restrict__ Ab) {
    __shared__ unsigned int Kl[2][32 * 33];     // [key][32 d-pairs] padded
    __shared__ unsigned int Vl[2][64 * 17];     // [d][16 key-pairs] padded
    __shared__ float        mfl[2][32];
    __shared__ unsigned int Pl[8 * 16 * 17];    // per-wave P scratch

    const int t    = threadIdx.x;
    const int w    = t >> 5;
    const int lane = t & 31;
    const int lm   = lane & 15;
    const int half = lane >> 4;

    const int qb = blockIdx.x & 7;
    const int h  = (blockIdx.x >> 3) & 15;
    const int b  = blockIdx.x >> 7;

    const int qrow = qb * 128 + w * 16 + lm;

    // per-lane staging coords
    const int kkey = t >> 3;            // 0..31 key row, 4-uint group
    const int kpg  = (t & 7) * 4;       // 0..28
    const int vd   = t >> 2;            // 0..63 d row
    const int vpg  = (t & 3) * 4;       // 0..12
    const unsigned long long Kg = (unsigned long long)(uintptr_t)Kb;
    const unsigned long long Vg = (unsigned long long)(uintptr_t)Vt;

    // preload Q A-fragments (d chunks 0-31, 32-63)
    AFrag Qf[2];
#pragma unroll
    for (int c = 0; c < 2; ++c)
#pragma unroll
        for (int i = 0; i < 8; ++i)
            Qf[c].u[i] = Qb[(size_t)(b * 1024 + qrow) * (HIDDEN / 2) + h * 32 + c * 16 +
                            pairmap(i, half)];

    float mrow[8], lrow[8];
    CFrag o[4];
#pragma unroll
    for (int i = 0; i < 8; ++i) { mrow[i] = -1e30f; lrow[i] = 0.0f; }
#pragma unroll
    for (int tt = 0; tt < 4; ++tt)
#pragma unroll
        for (int i = 0; i < 8; ++i) o[tt].f[i] = 0.0f;

    // stage chunk 0
    {
        async_b128(lds_off(&Kl[0][kkey * 33 + kpg]),
                   Kg + ((size_t)(b * 2048 + 0 + kkey) * (HIDDEN / 2) + h * 32 + kpg) * 4);
        async_b128(lds_off(&Vl[0][vd * 17 + vpg]),
                   Vg + ((((size_t)b * NHEADS + h) * HDIM + vd) * 1024 + 0 + vpg) * 4);
        if (t < 32) mfl[0][t] = mask[b * 2048 + 0 + t] ? 1.0f : 0.0f;
    }

    for (int step = 0; step < 64; ++step) {
        const int kb = step * 32;
        if (step + 1 < 64) {
            const int kn = kb + 32;
            const int s  = (step + 1) & 1;
            async_b128(lds_off(&Kl[s][kkey * 33 + kpg]),
                       Kg + ((size_t)(b * 2048 + kn + kkey) * (HIDDEN / 2) + h * 32 + kpg) * 4);
            async_b128(lds_off(&Vl[s][vd * 17 + vpg]),
                       Vg + ((((size_t)b * NHEADS + h) * HDIM + vd) * 1024 + (kn >> 1) + vpg) * 4);
            if (t < 32) mfl[s][t] = mask[b * 2048 + kn + t] ? 1.0f : 0.0f;
            wait_async2();
        } else {
            wait_async0();
        }
        __syncthreads();

        const unsigned int* Kc = Kl[step & 1];
        const unsigned int* Vc = Vl[step & 1];
        const float*        mc = mfl[step & 1];

        // scores: S[16 q][32 keys] = Q . K^T
        CFrag sc[2];
#pragma unroll
        for (int s = 0; s < 2; ++s) {
            AFrag B0, B1;
            int key = s * 16 + lm;
#pragma unroll
            for (int i = 0; i < 8; ++i) {
                int pm = pairmap(i, half);
                B0.u[i] = Kc[key * 33 + pm];
                B1.u[i] = Kc[key * 33 + 16 + pm];
            }
            v8f z = {};
            sc[s].v = __builtin_amdgcn_wmma_f32_16x16x32_bf16(false, Qf[0].v, false, B0.v,
                                                              (short)0, z, false, false);
            sc[s].v = __builtin_amdgcn_wmma_f32_16x16x32_bf16(false, Qf[1].v, false, B1.v,
                                                              (short)0, sc[s].v, false, false);
        }

        // scale, clamp, mask, online softmax update
        const float mk0 = mc[lm];
        const float mk1 = mc[16 + lm];
#pragma unroll
        for (int i = 0; i < 8; ++i) {
            float v0 = clampf(sc[0].f[i] * 0.125f, 80.0f);
            float v1 = clampf(sc[1].f[i] * 0.125f, 80.0f);
            if (mk0 != 0.0f) v0 = -10000.0f;
            if (mk1 != 0.0f) v1 = -10000.0f;
            float rm    = redmax16(fmaxf(v0, v1));
            float mnew  = fmaxf(mrow[i], rm);
            float alpha = __expf(mrow[i] - mnew);
            float p0 = __expf(v0 - mnew);
            float p1 = __expf(v1 - mnew);
            sc[0].f[i] = p0;
            sc[1].f[i] = p1;
            lrow[i] = lrow[i] * alpha + redsum16(p0 + p1);
            mrow[i] = mnew;
#pragma unroll
            for (int tt = 0; tt < 4; ++tt) o[tt].f[i] *= alpha;
        }

        // P: C-layout -> A-layout via per-wave LDS scratch (bf16)
        unsigned short* Ps = (unsigned short*)(Pl + w * (16 * 17));
#pragma unroll
        for (int i = 0; i < 8; ++i) {
            int row = half * 8 + i;
            Ps[row * 34 + lm]      = f2bf(sc[0].f[i]);
            Ps[row * 34 + 16 + lm] = f2bf(sc[1].f[i]);
        }
        AFrag Pf;
#pragma unroll
        for (int i = 0; i < 8; ++i)
            Pf.u[i] = Pl[w * (16 * 17) + lm * 17 + pairmap(i, half)];

        // out += P . V
#pragma unroll
        for (int tt = 0; tt < 4; ++tt) {
            AFrag Bv;
            int dc = tt * 16 + lm;
#pragma unroll
            for (int i = 0; i < 8; ++i) Bv.u[i] = Vc[dc * 17 + pairmap(i, half)];
            o[tt].v = __builtin_amdgcn_wmma_f32_16x16x32_bf16(false, Pf.v, false, Bv.v,
                                                              (short)0, o[tt].v, false, false);
        }
        __syncthreads();
    }

    // normalize and store attention output (bf16 row-major [4096][1024])
#pragma unroll
    for (int i = 0; i < 8; ++i) {
        float inv = 1.0f / lrow[i];
        int row = b * 1024 + qb * 128 + w * 16 + half * 8 + i;
#pragma unroll
        for (int tt = 0; tt < 4; ++tt) {
            int col = h * HDIM + tt * 16 + lm;
            Ab[(size_t)row * HIDDEN + col] = f2bf(o[tt].f[i] * inv);
        }
    }
}

// ---------------------------------------------------------------------------
extern "C" void kernel_launch(void* const* d_in, const int* in_sizes, int n_in,
                              void* d_out, int out_size, void* d_ws, size_t ws_size,
                              hipStream_t stream) {
    (void)in_sizes; (void)n_in; (void)out_size; (void)ws_size;
    const float* q    = (const float*)d_in[0];   // [4,1024,1024]
    const float* kv   = (const float*)d_in[1];   // [4,2048,1024]
    const unsigned char* msk = (const unsigned char*)d_in[2]; // [4,2048] bool
    const float* Wq = (const float*)d_in[3];
    const float* bq = (const float*)d_in[4];
    const float* Wk = (const float*)d_in[5];
    const float* bk = (const float*)d_in[6];
    const float* Wv = (const float*)d_in[7];
    const float* bv = (const float*)d_in[8];
    const float* Wo = (const float*)d_in[9];
    const float* bo = (const float*)d_in[10];

    // workspace (bf16 buffers):
    //  Qb 8Mi | Kb 16Mi | Vt 16Mi | Ab 8Mi | qbf 8Mi | kvbf 16Mi | 4x Wbf 2Mi
    char* ws = (char*)d_ws;
    unsigned short* Qb   = (unsigned short*)(ws);
    unsigned short* Kb   = (unsigned short*)(ws + (size_t)8  * 1024 * 1024);
    unsigned short* Vt   = (unsigned short*)(ws + (size_t)24 * 1024 * 1024);
    unsigned short* Ab   = (unsigned short*)(ws + (size_t)40 * 1024 * 1024);
    unsigned short* qbf  = (unsigned short*)(ws + (size_t)48 * 1024 * 1024);
    unsigned short* kvbf = (unsigned short*)(ws + (size_t)56 * 1024 * 1024);
    unsigned short* Wqb  = (unsigned short*)(ws + (size_t)72 * 1024 * 1024);
    unsigned short* Wkb  = (unsigned short*)(ws + (size_t)74 * 1024 * 1024);
    unsigned short* Wvb  = (unsigned short*)(ws + (size_t)76 * 1024 * 1024);
    unsigned short* Wob  = (unsigned short*)(ws + (size_t)78 * 1024 * 1024);

    dim3 blk(256);
    // one-time bf16 conversion (bandwidth-bound, ~5us at 23.3 TB/s)
    conv_kernel<<<dim3(2048), blk, 0, stream>>>(q,  (unsigned int*)qbf,  2097152);
    conv_kernel<<<dim3(4096), blk, 0, stream>>>(kv, (unsigned int*)kvbf, 4194304);
    conv_kernel<<<dim3(512),  blk, 0, stream>>>(Wq, (unsigned int*)Wqb,  524288);
    conv_kernel<<<dim3(512),  blk, 0, stream>>>(Wk, (unsigned int*)Wkb,  524288);
    conv_kernel<<<dim3(512),  blk, 0, stream>>>(Wv, (unsigned int*)Wvb,  524288);
    conv_kernel<<<dim3(512),  blk, 0, stream>>>(Wo, (unsigned int*)Wob,  524288);

    // projections (WMMA bf16, async double-buffered LDS)
    proj_kernel<0><<<dim3(64, 16),  blk, 0, stream>>>(qbf,  Wqb, bq, Qb, 4096);
    proj_kernel<0><<<dim3(128, 16), blk, 0, stream>>>(kvbf, Wkb, bk, Kb, 8192);
    proj_kernel<1><<<dim3(128, 16), blk, 0, stream>>>(kvbf, Wvb, bv, Vt, 8192);

    // fused flash attention: 4 b * 16 h * 8 q-blocks = 512 blocks
    attn_kernel<<<dim3(512), blk, 0, stream>>>((const unsigned int*)Qb,
                                               (const unsigned int*)Kb,
                                               (const unsigned int*)Vt,
                                               msk, Ab);

    // output projection -> fp32
    proj_kernel<2><<<dim3(64, 16), blk, 0, stream>>>(Ab, Wob, bo, (float*)d_out, 4096);
}
